// DDSP_69569880261056
// MI455X (gfx1250) — matrix operand order repaired and checked by reading, not compiled, not run on hardware
//
#include <hip/hip_runtime.h>
#include <hip/hip_bf16.h>
#include <math.h>

typedef _Float16 f16;
typedef __attribute__((ext_vector_type(16))) _Float16 v16h;
typedef __attribute__((ext_vector_type(8)))  float    v8f;
typedef __attribute__((ext_vector_type(4)))  unsigned int v4u;
typedef __attribute__((ext_vector_type(8)))  int v8i_;
typedef __attribute__((ext_vector_type(4)))  int v4i_;

#define BBATCH 32
#define TT 1000
#define HHID 512
#define NBANDS 65
#define BLKSZ 256
#define SRR 48000
#define MROWS (BBATCH*TT)        // 32000
#define SIGLEN (TT*BLKSZ)        // 256000
#define NPART 188                // ceil(48000/256)
#define NBLKS 1000               // 256-sample blocks per batch

// ---------------- device helpers ----------------
__device__ __forceinline__ float warp_sum(float v){
#pragma unroll
  for (int m = 16; m >= 1; m >>= 1) v += __shfl_xor(v, m, 32);
  return v;
}
__device__ __forceinline__ float sigm(float x){ return 1.f/(1.f+expf(-x)); }
__device__ __forceinline__ float scale_fn(float x){
  float s = sigm(x);
  return 2.f*powf(s, 2.302585092994046f) + 1e-7f;  // 2*sigmoid^ln(10)+1e-7
}
__device__ __forceinline__ v8f wmma16(v16h a, v16h b, v8f c){
  return __builtin_amdgcn_wmma_f32_16x16x32_f16(false, a, false, b, (short)0, c, false, false);
}
// A fragment: 16x32 f16 tile at tp (row-major, leading dim ld).  Per ISA 16-bit
// A layout: lane L holds row M=L%16; VGPR0-3 K=(L>=16?8:0)+2v+w, VGPR4-7 +16.
__device__ __forceinline__ v16h load_a_frag(const f16* tp, int ld, int lane){
  v16h a; const int m = lane & 15; const int hi = (lane & 16) ? 8 : 0;
#pragma unroll
  for (int h = 0; h < 16; ++h){
    int v = h >> 1, o = h & 1;
    int kk = (v < 4) ? (hi + 2*v + o) : (16 + hi + 2*(v-4) + o);
    a[h] = tp[m*ld + kk];
  }
  return a;
}
// B fragment from W [N,K] row-major: n in [nb,nb+16), k in [k0,k0+32).
// B layout matches A: lane half h holds K=8h+0..7 then 16+8h+0..7 (+k0), N=lane%16.
__device__ __forceinline__ v16h load_b_frag(const f16* W, int ldw, int nb, int k0, int lane){
  v16h b; const int n = nb + (lane & 15); const int kb = (lane & 16) ? 8 : 0;
  const f16* p = W + (size_t)n*ldw + k0 + kb;
#pragma unroll
  for (int h = 0; h < 16; ++h) b[h] = p[(h < 8) ? h : (h + 8)];
  return b;
}

// Tensor Data Mover: 1-D contiguous tile (2-byte elements, nelem <= 65535) from
// global memory into LDS at byte offset lds_byte_off.  D# per ISA 08_async_tensor:
// group0 = {count=1 | lds_addr | global_addr[56:0] | type=2}, group1 = {data_size=2B,
// tensor_dim0=tile_dim0=nelem, dims1+ trivial}.  clang-23 6-arg builtin
// (g0 u32x4, g1 i32x8, g2 i32x4, g3 i32x4, g4 i32x8, cpol).
__device__ __forceinline__ void tdm_load_1d(unsigned lds_byte_off, const void* gptr,
                                            unsigned nelem){
  unsigned long long ga = (unsigned long long)gptr;
  v4u g0;
  g0[0] = 1u;                                                 // count=1 (user D#)
  g0[1] = lds_byte_off;                                       // lds_addr
  g0[2] = (unsigned)ga;                                       // global_addr[31:0]
  g0[3] = (unsigned)((ga >> 32) & 0x1FFFFFFull) | (2u << 30); // addr[56:32] | type=2
  v8i_ g1;
  g1[0] = 0x10000;                                 // workgroup_mask=0, data_size=1 (2B)
  g1[1] = (int)((nelem & 0xFFFFu) << 16);          // tensor_dim0[15:0]  @ bits 63:48
  g1[2] = (int)(((nelem >> 16) & 0xFFFFu) | (1u << 16)); // dim0 hi | tensor_dim1=1
  g1[3] = (int)((nelem & 0xFFFFu) << 16);          // tile_dim0 @ bits 127:112
  g1[4] = 1;                                       // tile_dim1 = 1
  g1[5] = (int)nelem;                              // tensor_dim0_stride lo
  g1[6] = 0;
  g1[7] = 0;
  v4i_ gz4 = {0, 0, 0, 0};
  v8i_ gz8 = {0, 0, 0, 0, 0, 0, 0, 0};
  __builtin_amdgcn_tensor_load_to_lds(g0, g1, gz4, gz4, gz8, 0);
}

struct c2 { float x, y; };
__device__ __forceinline__ c2 cmul(c2 a, c2 b){ return c2{a.x*b.x - a.y*b.y, a.x*b.y + a.y*b.x}; }

// radix-2 Stockham FFT, N=512, 256 threads, ping-pong LDS buffers.
__device__ c2* fft512(c2* src, c2* dst, int tid){
  for (int l = 1; l < 512; l <<= 1){
    int grp = tid / l, k = tid - grp*l;
    float ang = -3.14159265358979f * (float)k / (float)l;
    c2 w{cosf(ang), sinf(ang)};
    c2 a = src[grp*l + k];
    c2 b = src[grp*l + k + 256];
    c2 bw = cmul(b, w);
    dst[grp*2*l + k]     = c2{a.x+bw.x, a.y+bw.y};
    dst[grp*2*l + k + l] = c2{a.x-bw.x, a.y-bw.y};
    __syncthreads();
    c2* t = src; src = dst; dst = t;
  }
  return src;
}

// ---------------- small prep kernels ----------------
__global__ void k_f32_to_f16_pad(f16* dst, const float* src, int N, int K, int Kp){
  int idx = blockIdx.x*256 + threadIdx.x;
  if (idx >= N*Kp) return;
  int n = idx / Kp, k = idx - n*Kp;
  dst[idx] = (k < K) ? (f16)src[(size_t)n*K + k] : (f16)0.f;
}
__global__ void k_build_ccos(f16* C){           // [128][96] irfft-as-GEMM matrix
  int idx = blockIdx.x*256 + threadIdx.x;
  if (idx >= 128*96) return;
  int n = idx / 96, k = idx - n*96;
  float v = 0.f;
  if (k < NBANDS){
    float c = cosf(6.283185307179586f * (float)k * (float)n / 128.f);
    float w = (k == 0 || k == NBANDS-1) ? 1.f : 2.f;
    v = w * c * (1.f/128.f);
  }
  C[idx] = (f16)v;
}
__global__ void k_conv_upsW(f16* w16, float* biasT, const float* upsW, const float* upsb){
  int idx = blockIdx.x*256 + threadIdx.x;
  if (idx < 4000) biasT[idx] = upsb[idx >> 2];
  if (idx >= 4000*1024) return;
  int n = idx >> 10, k = idx & 1023;            // transpose [i,(o,k)] -> [(o,k), i]
  w16[idx] = (k < TT) ? (f16)upsW[(size_t)k*4000 + n] : (f16)0.f;
}
__global__ void k_gamma_beta(const float* mh, const float* Wg, const float* bg,
                             const float* Wb, const float* bbt, float* gam, float* bet){
  int idx = blockIdx.x*256 + threadIdx.x;
  if (idx >= 32*1024) return;
  int b = idx >> 10, c = idx & 1023;
  float sg = bg[c], sb = bbt[c];
#pragma unroll
  for (int k = 0; k < 16; ++k){
    float m = mh[b*16 + k];
    sg += m * Wg[c*16 + k];
    sb += m * Wb[c*16 + k];
  }
  gam[idx] = sg; bet[idx] = sb;
}
// first MLP layer: scalar input -> 512 feats, LN + LeakyReLU (pure VALU, K=1)
__global__ void k_layer1(const float* __restrict__ x, const float* __restrict__ W,
                         const float* __restrict__ bv, const float* __restrict__ g,
                         const float* __restrict__ be, f16* __restrict__ out16){
  __shared__ float red[16];
  const int row = blockIdx.x, tid = threadIdx.x;
  float xv = x[row];
  float v0 = xv*W[tid]     + bv[tid];
  float v1 = xv*W[tid+256] + bv[tid+256];
  float s = v0 + v1, s2 = v0*v0 + v1*v1;
  s = warp_sum(s); s2 = warp_sum(s2);
  if ((tid & 31) == 0){ red[tid >> 5] = s; red[8 + (tid >> 5)] = s2; }
  __syncthreads();
  float S = 0.f, S2 = 0.f;
#pragma unroll
  for (int i = 0; i < 8; ++i){ S += red[i]; S2 += red[8+i]; }
  float mean = S*(1.f/512.f), var = S2*(1.f/512.f) - mean*mean;
  float inv = rsqrtf(var + 1e-5f);
  float y0 = (v0 - mean)*inv*g[tid]     + be[tid];       y0 = y0 >= 0.f ? y0 : 0.01f*y0;
  float y1 = (v1 - mean)*inv*g[tid+256] + be[tid+256];   y1 = y1 >= 0.f ? y1 : 0.01f*y1;
  out16[(size_t)row*512 + tid]       = (f16)y0;
  out16[(size_t)row*512 + tid + 256] = (f16)y1;
}

// ---------------- WMMA GEMM + LayerNorm + LeakyReLU (+ optional FiLM) --------
// 512 threads = 16 waves; block tile 16 x 512 (full row so LN is block-local).
__global__ void k_gemm_ln(const f16* __restrict__ A, int lda,
                          const f16* __restrict__ W, int K,
                          const float* __restrict__ bias,
                          const float* __restrict__ g, const float* __restrict__ be,
                          const float* __restrict__ fg, const float* __restrict__ fb,
                          f16* __restrict__ out16, int ld16, int co16){
  __shared__ __align__(16) f16 As[16*32];
  __shared__ float Cs[16*512];
  const int tid = threadIdx.x, lane = tid & 31, wv = tid >> 5;
  const int m0 = blockIdx.x * 16;
  v8f c0 = {}; v8f c1 = {};
  for (int k0 = 0; k0 < K; k0 += 32){
    __syncthreads();
    if (tid < 64){                         // b128 tile staging: 64 lanes x 16B = 1KB
      int r = tid >> 2, seg = tid & 3;
      *(uint4*)&As[r*32 + seg*8] = *(const uint4*)&A[(size_t)(m0 + r)*lda + k0 + seg*8];
    }
    __syncthreads();
    v16h a  = load_a_frag(As, 32, lane);
    v16h b0 = load_b_frag(W, K, wv*32,      k0, lane);
    v16h b1 = load_b_frag(W, K, wv*32 + 16, k0, lane);
    if (k0 + 32 < K)
      __builtin_prefetch(&W[(size_t)(wv*32 + (lane & 15))*K + k0 + 32], 0, 3);
    c0 = wmma16(a, b0, c0);
    c1 = wmma16(a, b1, c1);
  }
  { const int rbv = (lane & 16) ? 8 : 0;
    const int n0  = wv*32 + (lane & 15);
#pragma unroll
    for (int r = 0; r < 8; ++r){
      Cs[(rbv+r)*512 + n0]      = c0[r] + bias[n0];
      Cs[(rbv+r)*512 + n0 + 16] = c1[r] + bias[n0 + 16];
    } }
  __syncthreads();
  float xv[16]; float s = 0.f, s2 = 0.f;
#pragma unroll
  for (int j = 0; j < 16; ++j){
    float x = Cs[wv*512 + lane + 32*j];
    xv[j] = x; s += x; s2 += x*x;
  }
  s = warp_sum(s); s2 = warp_sum(s2);
  float mean = s * (1.f/512.f);
  float inv  = rsqrtf(s2*(1.f/512.f) - mean*mean + 1e-5f);
  const int row = m0 + wv;
  const int bb  = row / TT;
#pragma unroll
  for (int j = 0; j < 16; ++j){
    int n = lane + 32*j;
    float y = (xv[j] - mean) * inv * g[n] + be[n];
    y = (y >= 0.f) ? y : 0.01f*y;
    if (fg) y = fg[bb*1024 + co16 + n]*y + fb[bb*1024 + co16 + n];   // FiLM
    out16[(size_t)row*ld16 + co16 + n] = (f16)y;
  }
}

// ---------------- generic WMMA GEMM (bias, optional scale_fn) ----------------
// 256 threads = 8 waves; block tile 64x64; LDS-staged A and B (b128 per thread).
__global__ void k_gemm_bias(const f16* __restrict__ A, int lda, int M,
                            const f16* __restrict__ W, int ldw, int N,
                            const float* __restrict__ bias,
                            float* __restrict__ out32,
                            f16* __restrict__ out16, int ld16,
                            int act, float actshift, int K){
  __shared__ __align__(16) f16 As[64*32];
  __shared__ __align__(16) f16 Bs[64*32];
  const int tid = threadIdx.x, lane = tid & 31, wv = tid >> 5;
  const int mi = wv & 3, np = wv >> 2;
  const int m0 = blockIdx.y * 64, n0 = blockIdx.x * 64;
  v8f c0 = {}; v8f c1 = {};
  for (int k0 = 0; k0 < K; k0 += 32){
    __syncthreads();
    { int r = tid >> 2, seg = tid & 3;
      int row = m0 + r;
      uint4 va = make_uint4(0,0,0,0);
      if (row < M) va = *(const uint4*)&A[(size_t)row*lda + k0 + seg*8];
      *(uint4*)&As[r*32 + seg*8] = va;
      int n = n0 + r;
      uint4 vb = make_uint4(0,0,0,0);
      if (n < N) vb = *(const uint4*)&W[(size_t)n*ldw + k0 + seg*8];
      *(uint4*)&Bs[r*32 + seg*8] = vb;
    }
    __syncthreads();
    v16h a  = load_a_frag(&As[mi*16*32], 32, lane);
    v16h b0 = load_b_frag(Bs, 32, np*32,      0, lane);
    v16h b1 = load_b_frag(Bs, 32, np*32 + 16, 0, lane);
    c0 = wmma16(a, b0, c0);
    c1 = wmma16(a, b1, c1);
  }
  const int rbv = (lane & 16) ? 8 : 0;
  const int nA  = n0 + np*32 + (lane & 15);
#pragma unroll
  for (int r = 0; r < 8; ++r){
    int m = m0 + mi*16 + rbv + r;
    if (m >= M) continue;
    float v0 = c0[r] + ((bias && nA    < N) ? bias[nA]    : 0.f);
    float v1 = c1[r] + ((bias && nA+16 < N) ? bias[nA+16] : 0.f);
    if (act == 1){ v0 = scale_fn(v0 + actshift); v1 = scale_fn(v1 + actshift); }
    if (out32){
      if (nA    < N) out32[(size_t)m*N + nA]    = v0;
      if (nA+16 < N) out32[(size_t)m*N + nA+16] = v1;
    }
    if (out16){
      if (nA    < ld16) out16[(size_t)m*ld16 + nA]    = (nA    < N) ? (f16)v0 : (f16)0.f;
      if (nA+16 < ld16) out16[(size_t)m*ld16 + nA+16] = (nA+16 < N) ? (f16)v1 : (f16)0.f;
    }
  }
}

__global__ void k_fill_a544(f16* A, const float* spec, const float* loud){
  int idx = blockIdx.x*256 + threadIdx.x;
  if (idx >= MROWS*32) return;
  int row = idx >> 5, j = idx & 31;
  float v = (j == 0) ? spec[row] : (j == 1) ? loud[row] : 0.f;
  A[(size_t)row*544 + 512 + j] = (f16)v;
}

// ---------------- GRU: 16 WGPs, Wh slice resident in LDS (loaded by TDM), ----
// h exchanged through L2; cluster barrier when cluster-launched, atomic
// sense-barrier otherwise.  All LDS is dynamic so TDM LDS offsets are static.
__global__ void k_gru(const f16* __restrict__ xi16,    // [32000,1536]
                      const f16* __restrict__ Wh16,    // [1536,512]
                      const float* __restrict__ bh,    // [1536]
                      f16* __restrict__ hbuf16,        // [2][32*512] (zeroed)
                      float* __restrict__ hbuf32,      // [32*512]    (zeroed)
                      f16* __restrict__ A544,
                      int* __restrict__ bar){          // [1024]      (zeroed)
  extern __shared__ char smem[];
  f16*   WhS = (f16*)smem;                                    // 96*512 f16 @ 0
  f16*   hA  = (f16*)(smem + 96*512*2);                       // 32*512 f16 @ 98304
  float* ghs = (float*)(smem + 96*512*2 + 32*512*2);          // 32*96 f32
  float* bhS = (float*)(smem + 96*512*2 + 32*512*2 + 32*96*4);
  const int tid = threadIdx.x, lane = tid & 31, wv = tid >> 5;
  const int c0 = blockIdx.x * 32;          // this WG owns hidden cols [c0,c0+32)
  const bool in_cluster = (__builtin_amdgcn_cluster_id_x() != 0);
  // Wh slice: 3 contiguous 32-row x 512 panels (r/z/n) -> LDS via TDM
  if (wv == 0){
    tdm_load_1d(0u,     Wh16 + (size_t)(c0)*512,        16384u);
    tdm_load_1d(32768u, Wh16 + (size_t)(512  + c0)*512, 16384u);
    tdm_load_1d(65536u, Wh16 + (size_t)(1024 + c0)*512, 16384u);
    __builtin_amdgcn_s_wait_tensorcnt(0);
  }
  if (tid < 96){
    int rr = tid;
    int gr = (rr < 32) ? (c0 + rr) : (rr < 64) ? (512 + c0 + rr - 32)
                                               : (1024 + c0 + rr - 64);
    bhS[tid] = bh[gr];
  }
  __syncthreads();
  for (int t = 0; t < TT; ++t){
    const int rbk = t & 1;
    f16* hdst = hbuf16 + (size_t)(1 - rbk)*16384;
    __syncthreads();
    if (wv == 0){                          // stage h_prev (32KB) via TDM DMA
      tdm_load_1d(98304u, hbuf16 + (size_t)rbk*16384, 16384u);
      __builtin_amdgcn_s_wait_tensorcnt(0);
    }
    __syncthreads();
    // gh = h_prev @ WhS^T : 12 16x16 tiles over 8 waves
    for (int tile = wv; tile < 12; tile += 8){
      int mi = tile & 1, ni = tile >> 1;
      v8f c = {};
      for (int k0 = 0; k0 < 512; k0 += 32){
        v16h a = load_a_frag(&hA[(mi*16)*512 + k0], 512, lane);
        v16h b = load_b_frag(WhS, 512, ni*16, k0, lane);
        c = wmma16(a, b, c);
      }
      int rbv = (lane & 16) ? 8 : 0, nc = ni*16 + (lane & 15);
#pragma unroll
      for (int r = 0; r < 8; ++r) ghs[(mi*16 + rbv + r)*96 + nc] = c[r];
    }
    __syncthreads();
#pragma unroll
    for (int i = 0; i < 4; ++i){
      int e = tid + 256*i;
      int m = e >> 5, j = e & 31;
      int col = c0 + j;
      size_t row = (size_t)m*TT + t;
      float gi_r = (float)xi16[row*1536 + col];
      float gi_z = (float)xi16[row*1536 + 512 + col];
      float gi_n = (float)xi16[row*1536 + 1024 + col];
      float gr = ghs[m*96 + j]      + bhS[j];
      float gz = ghs[m*96 + 32 + j] + bhS[32 + j];
      float gn = ghs[m*96 + 64 + j] + bhS[64 + j];
      float r = sigm(gi_r + gr);
      float z = sigm(gi_z + gz);
      float n = tanhf(gi_n + r*gn);
      float hp = hbuf32[m*512 + col];
      float hn = (1.f - z)*n + z*hp;
      hbuf32[m*512 + col] = hn;
      hdst[m*512 + col]   = (f16)hn;
      A544[row*544 + col] = (f16)hn;
    }
    __threadfence();
    __syncthreads();
    if (in_cluster){
      __builtin_amdgcn_s_cluster_barrier();     // s_barrier_signal/wait -3
    } else {
      if (tid == 0){
        atomicAdd(&bar[t], 1);
        volatile int* vb = bar + t;
        while (*vb < 16) { }
      }
    }
    __syncthreads();
  }
}

// ---------------- proj1 (N=2, wave-per-row VALU dot) -------------------------
__global__ void k_proj1(const f16* __restrict__ h, const float* __restrict__ W,
                        const float* __restrict__ b, float* __restrict__ tp){
  int gw = (blockIdx.x*256 + threadIdx.x) >> 5;
  int lane = threadIdx.x & 31;
  if (gw >= MROWS) return;
  float a0 = 0.f, a1 = 0.f;
  for (int k = lane; k < 512; k += 32){
    float xv = (float)h[(size_t)gw*512 + k];
    a0 += xv*W[k]; a1 += xv*W[512 + k];
  }
  a0 = warp_sum(a0); a1 = warp_sum(a1);
  if (lane == 0){
    tp[(size_t)gw*2]     = scale_fn(a0 + b[0]);
    tp[(size_t)gw*2 + 1] = scale_fn(a1 + b[1]);
  }
}

// ---------------- per-(b,t) 512-pt FFT band-noise convolution ----------------
__global__ void k_fft_conv(const float* __restrict__ noise_u,
                           const float* __restrict__ ir, float* __restrict__ noise_out){
  __shared__ c2 bufA[512];
  __shared__ c2 bufB[512];
  const int row = blockIdx.x, tid = threadIdx.x;
  const int b = row / TT, t = row - b*TT;
  // pack s (right-padded) and impulse (left-padded, window/roll folded) as complex
#pragma unroll
  for (int i = 0; i < 2; ++i){
    int n = tid + 256*i;
    float sv = (n < 256) ? noise_u[(size_t)row*256 + n] : 0.f;
    float kv = 0.f;
    if (n >= 256){
      int j = n - 256;
      if (j < 64){
        float w = 0.5f - 0.5f*cosf(6.283185307179586f*(float)(j+64)/128.f);
        kv = w * ir[(size_t)row*128 + j];
      } else if (j >= 192){
        float w = 0.5f - 0.5f*cosf(6.283185307179586f*(float)(j-192)/128.f);
        kv = w * ir[(size_t)row*128 + j - 128];
      }
    }
    bufA[n] = c2{sv, kv};
  }
  __syncthreads();
  c2* Z = fft512(bufA, bufB, tid);
  c2* O = (Z == bufA) ? bufB : bufA;
#pragma unroll
  for (int i = 0; i < 2; ++i){
    int f = tid + 256*i;
    c2 zf = Z[f];
    c2 zc = Z[(512 - f) & 511]; zc.y = -zc.y;
    c2 S  = c2{0.5f*(zf.x + zc.x), 0.5f*(zf.y + zc.y)};
    c2 Kf = c2{0.5f*(zf.y - zc.y), -0.5f*(zf.x - zc.x)};
    c2 P  = cmul(S, Kf);
    O[f] = c2{P.x, -P.y};                    // conj for inverse-via-forward
  }
  __syncthreads();
  c2* tmp = (O == bufA) ? bufB : bufA;
  c2* Q = fft512(O, tmp, tid);
#pragma unroll
  for (int i = 0; i < 2; ++i){
    int n = tid + 256*i;
    if (n >= 256)
      noise_out[(size_t)b*SIGLEN + (size_t)t*256 + (n - 256)] = Q[n].x * (1.f/512.f);
  }
}

// ---------------- transient + signal sum -------------------------------------
__global__ void k_prep_tfreqA(const float* tp, f16* A){      // [32][1024]
  int idx = blockIdx.x*256 + threadIdx.x;
  if (idx >= 32*1024) return;
  int b = idx >> 10, i = idx & 1023;
  A[idx] = (i < TT) ? (f16)tp[((size_t)b*TT + i)*2 + 1] : (f16)0.f;
}
__global__ void k_transient_signal(const float* __restrict__ tfreq,
                                   const float* __restrict__ tp,
                                   const float* __restrict__ noise_out,
                                   float* __restrict__ tr_out, float* __restrict__ sig){
  size_t idx = (size_t)blockIdx.x*256 + threadIdx.x;
  if (idx >= (size_t)BBATCH*SIGLEN) return;
  int b = (int)(idx / SIGLEN);
  int i = (int)(idx - (size_t)b*SIGLEN);
  int j = i >> 6, nn = i & 63;                 // frame of 64 among 4000
  float f  = tfreq[(size_t)b*4000 + j];
  float ph = 6.283185307179586f * f * (float)(nn + 1);
  float amp = tp[((size_t)b*TT + (i >> 8))*2];
  float tr = amp * sinf(ph);
  tr_out[idx] = tr;
  sig[idx] = noise_out[idx] + tr;
}

// ---------------- reverb: partitioned FFT convolution ------------------------
__global__ void k_build_rev_imp(const float* rn, const float* dec, const float* wet,
                                float* imp){
  int idx = blockIdx.x*256 + threadIdx.x;
  if (idx >= NPART*256) return;
  float v = 0.f;
  if (idx < SRR){
    float sp = log1pf(expf(-dec[0]));          // softplus(-decay)
    float tg = (float)idx / (float)SRR;
    v = rn[idx] * expf(-sp * tg * 500.f) * sigm(wet[0]);
  }
  if (idx == 0) v = 1.f;
  imp[idx] = v;
}
__global__ void k_fft512_fwd(const float* __restrict__ src, c2* __restrict__ spec){
  __shared__ c2 bufA[512];
  __shared__ c2 bufB[512];
  const int blk = blockIdx.x, tid = threadIdx.x;
#pragma unroll
  for (int i = 0; i < 2; ++i){
    int n = tid + 256*i;
    bufA[n] = c2{ (n < 256) ? src[(size_t)blk*256 + n] : 0.f, 0.f };
  }
  __syncthreads();
  c2* Z = fft512(bufA, bufB, tid);
#pragma unroll
  for (int i = 0; i < 2; ++i){
    int n = tid + 256*i;
    spec[(size_t)blk*512 + n] = Z[n];
  }
}
__global__ void k_rev_accum(const c2* __restrict__ Sspec, const c2* __restrict__ Kspec,
                            float* __restrict__ ybuf){
  __shared__ c2 bufA[512];
  __shared__ c2 bufB[512];
  const int blk = blockIdx.x;
  const int b = blk / 125, m0 = (blk - b*125) * 8;
  const int tid = threadIdx.x;
  c2 acc0[8] = {}; c2 acc1[8] = {};
  for (int d = 0; d < NPART; ++d){
    c2 kk0 = Kspec[(size_t)d*512 + tid];
    c2 kk1 = Kspec[(size_t)d*512 + tid + 256];
#pragma unroll
    for (int mi = 0; mi < 8; ++mi){
      int j = m0 + mi - d;
      if ((unsigned)j < (unsigned)NBLKS){
        const c2* Sp = &Sspec[((size_t)b*NBLKS + j)*512];
        c2 p0 = cmul(Sp[tid], kk0);
        c2 p1 = cmul(Sp[tid + 256], kk1);
        acc0[mi].x += p0.x; acc0[mi].y += p0.y;
        acc1[mi].x += p1.x; acc1[mi].y += p1.y;
      }
    }
  }
  for (int mi = 0; mi < 8; ++mi){
    __syncthreads();
    bufA[tid]       = c2{acc0[mi].x, -acc0[mi].y};
    bufA[tid + 256] = c2{acc1[mi].x, -acc1[mi].y};
    __syncthreads();
    c2* Q = fft512(bufA, bufB, tid);
    float* yo = &ybuf[((size_t)b*NBLKS + m0 + mi)*512];
    yo[tid]       = Q[tid].x       * (1.f/512.f);
    yo[tid + 256] = Q[tid + 256].x * (1.f/512.f);
  }
}
__global__ void k_rev_final(const float* __restrict__ ybuf, float* __restrict__ out){
  size_t idx = (size_t)blockIdx.x*256 + threadIdx.x;
  if (idx >= (size_t)BBATCH*SIGLEN) return;
  int b = (int)(idx / SIGLEN);
  int i = (int)(idx - (size_t)b*SIGLEN);
  int m = i >> 8, n = i & 255;
  int mp = (m == 0) ? (NBLKS - 1) : (m - 1);
  out[idx] = ybuf[((size_t)b*NBLKS + m)*512 + n]
           + ybuf[((size_t)b*NBLKS + mp)*512 + 256 + n];
}

// ---------------- host driver ------------------------------------------------
static inline int ceilDiv(int a, int b){ return (a + b - 1) / b; }

enum {
  I_SPEC=0, I_LOUD=1, I_MH=2, I_NOISE=3,
  I_IN0=4, I_IN1=16,                    // 3 layers x {W,b,g,be}
  I_GRU_WI=28, I_GRU_WH=29, I_GRU_BI=30, I_GRU_BH=31,
  I_OUT=32,
  I_P0W=44, I_P0B=45, I_P1W=46, I_P1B=47,
  I_GAMW=48, I_GAMB=49, I_BETW=50, I_BETB=51,
  I_RNOISE=52, I_RDEC=53, I_RWET=54, I_UPSW=55, I_UPSB=56
};

extern "C" void kernel_launch(void* const* d_in, const int* in_sizes, int n_in,
                              void* d_out, int out_size, void* d_ws, size_t ws_size,
                              hipStream_t stream){
  (void)in_sizes; (void)n_in; (void)out_size; (void)ws_size;
  char* base = (char*)d_ws;
  size_t cur = 0;
  auto alloc = [&](size_t bytes)->char*{
    char* p = base + cur; cur += bytes; cur = (cur + 255) & ~(size_t)255; return p;
  };
  auto fin = [&](int i)->const float*{ return (const float*)d_in[i]; };

  // zeroed head: barrier slots, GRU h double-buffer (f16) and h state (f32)
  int*   bar    = (int*)  alloc(1024*sizeof(int));
  f16*   hbuf16 = (f16*)  alloc((size_t)2*32*512*sizeof(f16));
  float* hbuf32 = (float*)alloc((size_t)32*512*sizeof(float));
  size_t zeroBytes = cur;
  (void)hipMemsetAsync(d_ws, 0, zeroBytes, stream);

  // f16 weights
  f16* in0W2 = (f16*)alloc((size_t)512*512*2);
  f16* in0W3 = (f16*)alloc((size_t)512*512*2);
  f16* in1W2 = (f16*)alloc((size_t)512*512*2);
  f16* in1W3 = (f16*)alloc((size_t)512*512*2);
  f16* Wi16  = (f16*)alloc((size_t)1536*1024*2);
  f16* Wh16  = (f16*)alloc((size_t)1536*512*2);
  f16* outW1 = (f16*)alloc((size_t)512*544*2);
  f16* outW2 = (f16*)alloc((size_t)512*512*2);
  f16* outW3 = (f16*)alloc((size_t)512*512*2);
  f16* p0W   = (f16*)alloc((size_t)65*512*2);
  f16* Ccos  = (f16*)alloc((size_t)128*96*2);
  f16* upsW16= (f16*)alloc((size_t)4000*1024*2);
  float* biasT = (float*)alloc((size_t)4000*4);
  float* gam = (float*)alloc((size_t)32*1024*4);
  float* bet = (float*)alloc((size_t)32*1024*4);
  // activations / scratch
  f16* bufA   = (f16*)alloc((size_t)MROWS*512*2);
  f16* bufB   = (f16*)alloc((size_t)MROWS*512*2);
  f16* hcat16 = (f16*)alloc((size_t)MROWS*1024*2);
  f16* xi16   = (f16*)alloc((size_t)MROWS*1536*2);
  f16* A544   = (f16*)alloc((size_t)MROWS*544*2);
  f16* amp16  = (f16*)alloc((size_t)MROWS*96*2);
  float* ir   = (float*)alloc((size_t)MROWS*128*4);
  float* tp_s = (float*)alloc((size_t)MROWS*2*4);
  f16* A16t   = (f16*)alloc((size_t)32*1024*2);
  float* tfreq= (float*)alloc((size_t)32*4000*4);
  float* impb = (float*)alloc((size_t)NPART*256*4);
  c2*   Kspec = (c2*)  alloc((size_t)NPART*512*sizeof(c2));
  float* sig  = (float*)alloc((size_t)BBATCH*SIGLEN*4);
  c2*   Sspec = (c2*)  alloc((size_t)MROWS*512*sizeof(c2));
  float* ybuf = (float*)alloc((size_t)MROWS*512*4);

  float* out_sig   = (float*)d_out;
  float* out_noise = out_sig   + (size_t)BBATCH*SIGLEN;
  float* out_tr    = out_noise + (size_t)BBATCH*SIGLEN;

  // ---- weight conversions ----
  k_f32_to_f16_pad<<<ceilDiv(512*512,256),256,0,stream>>>(in0W2, fin(I_IN0+4), 512,512,512);
  k_f32_to_f16_pad<<<ceilDiv(512*512,256),256,0,stream>>>(in0W3, fin(I_IN0+8), 512,512,512);
  k_f32_to_f16_pad<<<ceilDiv(512*512,256),256,0,stream>>>(in1W2, fin(I_IN1+4), 512,512,512);
  k_f32_to_f16_pad<<<ceilDiv(512*512,256),256,0,stream>>>(in1W3, fin(I_IN1+8), 512,512,512);
  k_f32_to_f16_pad<<<ceilDiv(1536*1024,256),256,0,stream>>>(Wi16, fin(I_GRU_WI),1536,1024,1024);
  k_f32_to_f16_pad<<<ceilDiv(1536*512,256),256,0,stream>>>(Wh16, fin(I_GRU_WH),1536,512,512);
  k_f32_to_f16_pad<<<ceilDiv(512*544,256),256,0,stream>>>(outW1, fin(I_OUT+0), 512,514,544);
  k_f32_to_f16_pad<<<ceilDiv(512*512,256),256,0,stream>>>(outW2, fin(I_OUT+4), 512,512,512);
  k_f32_to_f16_pad<<<ceilDiv(512*512,256),256,0,stream>>>(outW3, fin(I_OUT+8), 512,512,512);
  k_f32_to_f16_pad<<<ceilDiv(65*512,256),256,0,stream>>>(p0W, fin(I_P0W), 65,512,512);
  k_build_ccos<<<ceilDiv(128*96,256),256,0,stream>>>(Ccos);
  k_conv_upsW<<<ceilDiv(4000*1024,256),256,0,stream>>>(upsW16, biasT, fin(I_UPSW), fin(I_UPSB));
  k_gamma_beta<<<ceilDiv(32*1024,256),256,0,stream>>>(fin(I_MH), fin(I_GAMW), fin(I_GAMB),
                                                      fin(I_BETW), fin(I_BETB), gam, bet);
  // ---- input MLPs + FiLM fused into layer-3 epilogue ----
  k_layer1<<<MROWS,256,0,stream>>>(fin(I_SPEC), fin(I_IN0+0), fin(I_IN0+1),
                                   fin(I_IN0+2), fin(I_IN0+3), bufA);
  k_gemm_ln<<<MROWS/16,512,0,stream>>>(bufA,512, in0W2,512, fin(I_IN0+5),
        fin(I_IN0+6), fin(I_IN0+7), nullptr,nullptr, bufB,512,0);
  k_gemm_ln<<<MROWS/16,512,0,stream>>>(bufB,512, in0W3,512, fin(I_IN0+9),
        fin(I_IN0+10), fin(I_IN0+11), gam,bet, hcat16,1024,0);
  k_layer1<<<MROWS,256,0,stream>>>(fin(I_LOUD), fin(I_IN1+0), fin(I_IN1+1),
                                   fin(I_IN1+2), fin(I_IN1+3), bufA);
  k_gemm_ln<<<MROWS/16,512,0,stream>>>(bufA,512, in1W2,512, fin(I_IN1+5),
        fin(I_IN1+6), fin(I_IN1+7), nullptr,nullptr, bufB,512,0);
  k_gemm_ln<<<MROWS/16,512,0,stream>>>(bufB,512, in1W3,512, fin(I_IN1+9),
        fin(I_IN1+10), fin(I_IN1+11), gam,bet, hcat16,1024,512);
  // ---- GRU input projection xi = hcat @ Wi^T + bi ----
  k_gemm_bias<<<dim3(1536/64, MROWS/64),256,0,stream>>>(hcat16,1024,MROWS, Wi16,1024,1536,
        fin(I_GRU_BI), nullptr, xi16,1536, 0,0.f, 1024);
  k_fill_a544<<<ceilDiv(MROWS*32,256),256,0,stream>>>(A544, fin(I_SPEC), fin(I_LOUD));
  // ---- GRU: 16 WGs, Wh resident in LDS via TDM, barrier per step ----
  const int SMEM_GRU = 96*512*2 + 32*512*2 + 32*96*4 + 96*4;
  k_gru<<<16,256,SMEM_GRU,stream>>>(xi16, Wh16, fin(I_GRU_BH), hbuf16, hbuf32, A544, bar);
  // ---- output MLP ----
  k_gemm_ln<<<MROWS/16,512,0,stream>>>(A544,544, outW1,544, fin(I_OUT+1),
        fin(I_OUT+2), fin(I_OUT+3), nullptr,nullptr, bufA,512,0);
  k_gemm_ln<<<MROWS/16,512,0,stream>>>(bufA,512, outW2,512, fin(I_OUT+5),
        fin(I_OUT+6), fin(I_OUT+7), nullptr,nullptr, bufB,512,0);
  k_gemm_ln<<<MROWS/16,512,0,stream>>>(bufB,512, outW3,512, fin(I_OUT+9),
        fin(I_OUT+10), fin(I_OUT+11), nullptr,nullptr, bufA,512,0);
  // ---- noise branch: band amps -> impulse (irfft-as-GEMM) -> FFT convolve ----
  k_gemm_bias<<<dim3(2, MROWS/64),256,0,stream>>>(bufA,512,MROWS, p0W,512,65,
        fin(I_P0B), nullptr, amp16,96, 1,-5.f, 512);
  k_gemm_bias<<<dim3(2, MROWS/64),256,0,stream>>>(amp16,96,MROWS, Ccos,96,128,
        nullptr, ir, nullptr,0, 0,0.f, 96);
  k_proj1<<<ceilDiv(MROWS*32,256),256,0,stream>>>(bufA, fin(I_P1W), fin(I_P1B), tp_s);
  k_fft_conv<<<MROWS,256,0,stream>>>(fin(I_NOISE), ir, out_noise);
  // ---- transient branch: ConvT1d-as-GEMM + sinusoid synth ----
  k_prep_tfreqA<<<ceilDiv(32*1024,256),256,0,stream>>>(tp_s, A16t);
  k_gemm_bias<<<dim3(ceilDiv(4000,64),1),256,0,stream>>>(A16t,1024,32, upsW16,1024,4000,
        biasT, tfreq, nullptr,0, 0,0.f, 1024);
  k_transient_signal<<<ceilDiv(BBATCH*SIGLEN,256),256,0,stream>>>(tfreq, tp_s,
        out_noise, out_tr, sig);
  // ---- reverb: partitioned FFT convolution ----
  k_build_rev_imp<<<ceilDiv(NPART*256,256),256,0,stream>>>(fin(I_RNOISE), fin(I_RDEC),
        fin(I_RWET), impb);
  k_fft512_fwd<<<NPART,256,0,stream>>>(impb, Kspec);
  k_fft512_fwd<<<MROWS,256,0,stream>>>(sig, Sspec);
  k_rev_accum<<<BBATCH*125,256,0,stream>>>(Sspec, Kspec, ybuf);
  k_rev_final<<<ceilDiv(BBATCH*SIGLEN,256),256,0,stream>>>(ybuf, out_sig);
}
